// GRU_54924041781711
// MI455X (gfx1250) — compile-verified, matrix operand order checked by reading, not run on hardware
//
#include <hip/hip_runtime.h>
#include <hip/hip_bf16.h>

// ---------------------------------------------------------------------------
// GRU LM forward for MI455X (gfx1250, wave32):
//   bf16 WMMA 16x16x32 GEMMs + Tensor Data Mover (TDM) async LDS staging.
// ---------------------------------------------------------------------------

typedef __bf16    v16bf __attribute__((ext_vector_type(16)));
typedef __bf16    v8bf  __attribute__((ext_vector_type(8)));
typedef float     v8f   __attribute__((ext_vector_type(8)));
typedef unsigned  v4u   __attribute__((ext_vector_type(4)));
typedef int       v8i   __attribute__((ext_vector_type(8)));
typedef int       v4i   __attribute__((ext_vector_type(4)));

constexpr int L_ = 2;
constexpr int H_ = 1024;
constexpr int E_ = 1024;
constexpr int T_ = 64;
constexpr int B_ = 128;
constexpr int V_ = 10000;
constexpr int KC = 32;           // K per WMMA chunk (bf16 16x16x32)
constexpr int NCH = H_ / KC;     // 32 K-chunks

#if defined(__HIP_DEVICE_COMPILE__) && __has_builtin(__builtin_amdgcn_tensor_load_to_lds) && \
    __has_builtin(__builtin_amdgcn_s_wait_tensorcnt)
#define USE_TDM 1
#else
#define USE_TDM 0
#endif

// --- WMMA helpers -----------------------------------------------------------

__device__ __forceinline__ v8f wmma_bf16(v16bf a, v16bf b, v8f c) {
    return __builtin_amdgcn_wmma_f32_16x16x32_bf16(
        /*neg_a=*/false, a, /*neg_b=*/false, b,
        /*c_mod=*/(short)0, c, /*reuse_a=*/false, /*reuse_b=*/false);
}

// 16x32 bf16 fragment from an LDS tile with row stride KC.
__device__ __forceinline__ v16bf lds_frag(const __bf16* tile, int rowBase, int lane) {
    int r  = rowBase + (lane & 15);
    int kb = (lane >> 4) << 3;
    const __bf16* p = tile + r * KC + kb;
    v8bf lo = *(const v8bf*)(p);
    v8bf hi = *(const v8bf*)(p + 16);
    return __builtin_shufflevector(lo, hi, 0,1,2,3,4,5,6,7,8,9,10,11,12,13,14,15);
}

// --- TDM helpers ------------------------------------------------------------
// Low 32 bits of a generic pointer to __shared__ == LDS byte offset.
__device__ __forceinline__ unsigned lds_off_of(const void* p) {
    return (unsigned)(unsigned long long)p;
}

// Issue one TDM 2D tile load: `rows` x `cols` elements of 2 bytes, global row
// stride `stride_elems`, rows past `tensor_rows` zero-filled (OOB semantics).
// D# bit layout per CDNA5 ISA ch.8 (group0: count/lds/global/type, group1:
// data_size, tensor_dim0/1, tile_dim0/1, dim0 stride).
// This toolchain's builtin is the 6-arg form:
//   (uint32x4 g0, int32x8 g1, int32x4 g2, int32x4 g3, int32x8 pad, i32 cpol)
__device__ __forceinline__ void tdm_load_2d(unsigned lds_byte_off, const void* g,
                                            unsigned cols, unsigned rows,
                                            unsigned tensor_rows, unsigned stride_elems) {
#if USE_TDM
    unsigned long long ga = (unsigned long long)g;
    v4u g0;
    g0[0] = 1u;                                                    // count=1, user D#
    g0[1] = lds_byte_off;                                          // lds_addr
    g0[2] = (unsigned)ga;                                          // global_addr[31:0]
    g0[3] = (unsigned)((ga >> 32) & 0x01FFFFFFu) | (2u << 30);     // addr[56:32] | type=2
    v8i g1;
    g1[0] = (int)(1u << 16);                                       // data_size = 2 bytes
    g1[1] = (int)((cols & 0xFFFFu) << 16);                         // tensor_dim0[15:0]
    g1[2] = (int)((cols >> 16) | ((tensor_rows & 0xFFFFu) << 16)); // dim0 hi | tensor_dim1 lo
    g1[3] = (int)((tensor_rows >> 16) | ((cols & 0xFFFFu) << 16)); // dim1 hi | tile_dim0
    g1[4] = (int)(rows & 0xFFFFu);                                 // tile_dim1 (tile_dim2=0)
    g1[5] = (int)stride_elems;                                     // tensor_dim0_stride lo
    g1[6] = 0;
    g1[7] = 0;
    v4i z4 = {0, 0, 0, 0};
    v8i z8 = {0, 0, 0, 0, 0, 0, 0, 0};
    __builtin_amdgcn_tensor_load_to_lds(g0, g1, z4, z4, z8, 0);
#else
    (void)lds_byte_off; (void)g; (void)cols; (void)rows;
    (void)tensor_rows; (void)stride_elems;
#endif
}

__device__ __forceinline__ void tdm_wait(int pending) {
#if USE_TDM
    if (pending) __builtin_amdgcn_s_wait_tensorcnt(7);
    else         __builtin_amdgcn_s_wait_tensorcnt(0);
#else
    (void)pending;
#endif
}

// --- Elementwise / prep kernels --------------------------------------------

__global__ void f2bf_kernel(const float* __restrict__ s, __bf16* __restrict__ d, size_t n) {
    size_t i = (size_t)blockIdx.x * blockDim.x + threadIdx.x;
    if (i < n) d[i] = (__bf16)s[i];
}

__global__ void embed_kernel(const int* __restrict__ tok, const float* __restrict__ emb,
                             __bf16* __restrict__ xe) {
    size_t i = (size_t)blockIdx.x * blockDim.x + threadIdx.x;
    if (i < (size_t)T_ * B_ * E_) {
        size_t tb = i / E_;
        int   e   = (int)(i - tb * E_);
        xe[i] = (__bf16)emb[(size_t)tok[tb] * E_ + e];
    }
}

__global__ void inith_kernel(const float* __restrict__ h0, float* __restrict__ hf,
                             __bf16* __restrict__ hb, int n) {
    int i = blockIdx.x * blockDim.x + threadIdx.x;
    if (i < n) { float v = h0[i]; hf[i] = v; hb[i] = (__bf16)v; }
}

__global__ void copyh_kernel(const float* __restrict__ hf, float* __restrict__ out, int n) {
    int i = blockIdx.x * blockDim.x + threadIdx.x;
    if (i < n) out[i] = hf[i];
}

// --- GRU cell, phase A ------------------------------------------------------
// accR = x@Wr^T + h@Ur^T ; accZ = x@Wz^T + h@Uz^T ; accC = x@Wh^T
// Block tile: M=128 (batch), N=32.  8 waves -> wave tile 16x32.
__global__ __launch_bounds__(256) void gru_cell_a(
    const __bf16* __restrict__ x,   // [B_,H_] bf16 layer input
    const __bf16* __restrict__ hb,  // [B_,H_] bf16 previous hidden
    const float*  __restrict__ hf,  // [B_,H_] f32 previous hidden
    const __bf16* __restrict__ Wr, const __bf16* __restrict__ Wz, const __bf16* __restrict__ Wh,
    const __bf16* __restrict__ Ur, const __bf16* __restrict__ Uz,
    const float* __restrict__ br, const float* __restrict__ bz,
    float* __restrict__ zout, float* __restrict__ cpart, __bf16* __restrict__ rhout)
{
    constexpr int NT = 32;
    __shared__ __align__(16) __bf16 sx[2][B_ * KC];
    __shared__ __align__(16) __bf16 sh[2][B_ * KC];
    __shared__ __align__(16) __bf16 sw[2][5][NT * KC];

    const int tid  = threadIdx.x;
    const int lane = tid & 31;
    const int wave = tid >> 5;
    const int wm = wave * 16;           // M offset inside 128
    const int n0 = blockIdx.x * NT;     // block N base

    const __bf16* wmat[5] = {Wr, Wz, Wh, Ur, Uz};

    v8f accR[2] = {};
    v8f accZ[2] = {};
    v8f accC[2] = {};

#if USE_TDM
    auto issue = [&](int kc, int p) {
        int k0 = kc * KC;
        tdm_load_2d(lds_off_of(&sx[p][0]), x + k0, KC, B_, B_, H_);
        tdm_load_2d(lds_off_of(&sh[p][0]), hb + k0, KC, B_, B_, H_);
        #pragma unroll
        for (int wI = 0; wI < 5; ++wI)
            tdm_load_2d(lds_off_of(&sw[p][wI][0]),
                        wmat[wI] + (size_t)n0 * H_ + k0, KC, NT, NT, H_);
    };
    if (tid < 32) issue(0, 0);
#endif

    for (int kc = 0; kc < NCH; ++kc) {
        const int p = kc & 1;
#if USE_TDM
        if (tid < 32) {
            if (kc + 1 < NCH) { issue(kc + 1, p ^ 1); tdm_wait(1); }
            else              { tdm_wait(0); }
        }
#else
        // manual staging fallback
        const int k0 = kc * KC;
        #pragma unroll
        for (int j = 0; j < 2; ++j) {
            int chunk = (tid << 1) | j;      // 0..511 over 128x32
            int row = chunk >> 2;
            int c8  = (chunk & 3) << 3;
            *(uint4*)(&sx[p][row * KC + c8]) = *(const uint4*)(&x [(size_t)row * H_ + k0 + c8]);
            *(uint4*)(&sh[p][row * KC + c8]) = *(const uint4*)(&hb[(size_t)row * H_ + k0 + c8]);
        }
        #pragma unroll
        for (int j = 0; j < 3; ++j) {
            int idx = tid + j * 256;         // 640 chunks over 5x(32x32)
            if (idx < 5 * NT * 4) {
                int wI  = idx >> 7;
                int rem = idx & 127;
                int row = rem >> 2;
                int c8  = (rem & 3) << 3;
                *(uint4*)(&sw[p][wI][row * KC + c8]) =
                    *(const uint4*)(&wmat[wI][(size_t)(n0 + row) * H_ + k0 + c8]);
            }
        }
#endif
        __syncthreads();

        v16bf ax = lds_frag(sx[p], wm, lane);
        v16bf ah = lds_frag(sh[p], wm, lane);
        #pragma unroll
        for (int ni = 0; ni < 2; ++ni) {
            v16bf bR  = lds_frag(sw[p][0], ni * 16, lane);
            v16bf bZ  = lds_frag(sw[p][1], ni * 16, lane);
            v16bf bH  = lds_frag(sw[p][2], ni * 16, lane);
            v16bf bUr = lds_frag(sw[p][3], ni * 16, lane);
            v16bf bUz = lds_frag(sw[p][4], ni * 16, lane);
            accR[ni] = wmma_bf16(ax, bR,  accR[ni]);
            accR[ni] = wmma_bf16(ah, bUr, accR[ni]);
            accZ[ni] = wmma_bf16(ax, bZ,  accZ[ni]);
            accZ[ni] = wmma_bf16(ah, bUz, accZ[ni]);
            accC[ni] = wmma_bf16(ax, bH,  accC[ni]);
        }
        __syncthreads();
    }

    // epilogue: lane holds col = lane&15, rows i + 8*(lane>=16)
    #pragma unroll
    for (int ni = 0; ni < 2; ++ni) {
        int col = n0 + ni * 16 + (lane & 15);
        int rb  = wm + ((lane >> 4) << 3);
        float brv = br[col];
        float bzv = bz[col];
        #pragma unroll
        for (int i = 0; i < 8; ++i) {
            size_t idx = (size_t)(rb + i) * H_ + col;
            float rg = 1.0f / (1.0f + expf(-(accR[ni][i] + brv)));
            float zg = 1.0f / (1.0f + expf(-(accZ[ni][i] + bzv)));
            zout[idx]  = zg;
            cpart[idx] = accC[ni][i];
            rhout[idx] = (__bf16)(rg * hf[idx]);
        }
    }
}

// --- GRU cell, phase B ------------------------------------------------------
// acc = (r*h)@Uh^T ; th = tanh(acc + cpart + bh) ; h' = (1-z)h + z*th
__global__ __launch_bounds__(256) void gru_cell_b(
    const __bf16* __restrict__ rh,  // [B_,H_] bf16
    const __bf16* __restrict__ Uh,  // [H_,H_] bf16
    const float* __restrict__ bh,
    const float* __restrict__ cpart,
    const float* __restrict__ zbuf,
    float* __restrict__ hf, __bf16* __restrict__ hbo,
    __bf16* __restrict__ topsOut)   // nullptr unless top layer
{
    constexpr int NT = 32;
    __shared__ __align__(16) __bf16 sa[2][B_ * KC];
    __shared__ __align__(16) __bf16 sb[2][NT * KC];

    const int tid  = threadIdx.x;
    const int lane = tid & 31;
    const int wave = tid >> 5;
    const int wm = wave * 16;
    const int n0 = blockIdx.x * NT;

    v8f acc[2] = {};

#if USE_TDM
    auto issue = [&](int kc, int p) {
        int k0 = kc * KC;
        tdm_load_2d(lds_off_of(&sa[p][0]), rh + k0, KC, B_, B_, H_);
        tdm_load_2d(lds_off_of(&sb[p][0]), Uh + (size_t)n0 * H_ + k0, KC, NT, NT, H_);
    };
    if (tid < 32) issue(0, 0);
#endif

    for (int kc = 0; kc < NCH; ++kc) {
        const int p = kc & 1;
#if USE_TDM
        if (tid < 32) {
            if (kc + 1 < NCH) { issue(kc + 1, p ^ 1); tdm_wait(1); }
            else              { tdm_wait(0); }
        }
#else
        const int k0 = kc * KC;
        #pragma unroll
        for (int j = 0; j < 2; ++j) {
            int chunk = (tid << 1) | j;
            int row = chunk >> 2;
            int c8  = (chunk & 3) << 3;
            *(uint4*)(&sa[p][row * KC + c8]) = *(const uint4*)(&rh[(size_t)row * H_ + k0 + c8]);
        }
        if (tid < NT * 4) {
            int row = tid >> 2;
            int c8  = (tid & 3) << 3;
            *(uint4*)(&sb[p][row * KC + c8]) =
                *(const uint4*)(&Uh[(size_t)(n0 + row) * H_ + k0 + c8]);
        }
#endif
        __syncthreads();

        v16bf a  = lds_frag(sa[p], wm, lane);
        v16bf b0 = lds_frag(sb[p], 0,  lane);
        v16bf b1 = lds_frag(sb[p], 16, lane);
        acc[0] = wmma_bf16(a, b0, acc[0]);
        acc[1] = wmma_bf16(a, b1, acc[1]);
        __syncthreads();
    }

    #pragma unroll
    for (int ni = 0; ni < 2; ++ni) {
        int col = n0 + ni * 16 + (lane & 15);
        int rb  = wm + ((lane >> 4) << 3);
        float bhv = bh[col];
        #pragma unroll
        for (int i = 0; i < 8; ++i) {
            size_t idx = (size_t)(rb + i) * H_ + col;
            float th = tanhf(acc[ni][i] + cpart[idx] + bhv);
            float z  = zbuf[idx];
            float hv = hf[idx];
            float hn = (1.0f - z) * hv + z * th;
            hf[idx] = hn;
            __bf16 hnb = (__bf16)hn;
            hbo[idx] = hnb;
            if (topsOut) topsOut[idx] = hnb;
        }
    }
}

// --- Output projection: logits = tops @ Wout^T + bout -----------------------
// M = T*B = 8192, N = 10000 (TDM zero-fills OOB rows), K = 1024.
// Block tile 128x64, 8 waves -> 4(M) x 2(N) of 32x32.
__global__ __launch_bounds__(256) void logits_gemm(
    const __bf16* __restrict__ A,    // [T_*B_, H_] bf16
    const __bf16* __restrict__ Wv,   // [V_, H_] bf16
    const float* __restrict__ bout,
    float* __restrict__ out)         // [T_*B_, V_] f32
{
    __shared__ __align__(16) __bf16 sa[2][128 * KC];
    __shared__ __align__(16) __bf16 sb[2][64 * KC];

    const int tid  = threadIdx.x;
    const int lane = tid & 31;
    const int wave = tid >> 5;
    const int wm = (wave & 3) * 32;
    const int wn = (wave >> 2) * 32;
    const int n0 = blockIdx.x * 64;
    const int m0 = blockIdx.y * 128;

    v8f acc[2][2] = {};

#if USE_TDM
    auto issue = [&](int kc, int p) {
        int k0 = kc * KC;
        tdm_load_2d(lds_off_of(&sa[p][0]), A + (size_t)m0 * H_ + k0, KC, 128, 128, H_);
        tdm_load_2d(lds_off_of(&sb[p][0]), Wv + (size_t)n0 * H_ + k0, KC, 64,
                    (unsigned)(V_ - n0), H_);   // rows past vocab end -> zero fill
    };
    if (tid < 32) issue(0, 0);
#endif

    for (int kc = 0; kc < NCH; ++kc) {
        const int p = kc & 1;
#if USE_TDM
        if (tid < 32) {
            if (kc + 1 < NCH) { issue(kc + 1, p ^ 1); tdm_wait(1); }
            else              { tdm_wait(0); }
        }
#else
        const int k0 = kc * KC;
        #pragma unroll
        for (int j = 0; j < 2; ++j) {
            int chunk = (tid << 1) | j;
            int row = chunk >> 2;
            int c8  = (chunk & 3) << 3;
            *(uint4*)(&sa[p][row * KC + c8]) =
                *(const uint4*)(&A[(size_t)(m0 + row) * H_ + k0 + c8]);
        }
        {
            int row = tid >> 2;
            int c8  = (tid & 3) << 3;
            int n   = n0 + row;
            uint4 v = make_uint4(0u, 0u, 0u, 0u);
            if (n < V_) v = *(const uint4*)(&Wv[(size_t)n * H_ + k0 + c8]);
            *(uint4*)(&sb[p][row * KC + c8]) = v;
        }
#endif
        __syncthreads();

        v16bf a0 = lds_frag(sa[p], wm,      lane);
        v16bf a1 = lds_frag(sa[p], wm + 16, lane);
        v16bf b0 = lds_frag(sb[p], wn,      lane);
        v16bf b1 = lds_frag(sb[p], wn + 16, lane);
        acc[0][0] = wmma_bf16(a0, b0, acc[0][0]);
        acc[0][1] = wmma_bf16(a0, b1, acc[0][1]);
        acc[1][0] = wmma_bf16(a1, b0, acc[1][0]);
        acc[1][1] = wmma_bf16(a1, b1, acc[1][1]);
        __syncthreads();
    }

    #pragma unroll
    for (int mi = 0; mi < 2; ++mi) {
        #pragma unroll
        for (int ni = 0; ni < 2; ++ni) {
            int col = n0 + wn + ni * 16 + (lane & 15);
            int rb  = m0 + wm + mi * 16 + ((lane >> 4) << 3);
            if (col < V_) {
                float bv = bout[col];
                #pragma unroll
                for (int i = 0; i < 8; ++i)
                    out[(size_t)(rb + i) * V_ + col] = acc[mi][ni][i] + bv;
            }
        }
    }
}

// ---------------------------------------------------------------------------

extern "C" void kernel_launch(void* const* d_in, const int* in_sizes, int n_in,
                              void* d_out, int out_size, void* d_ws, size_t ws_size,
                              hipStream_t stream) {
    (void)in_sizes; (void)n_in; (void)out_size; (void)ws_size;

    const int*   inputs = (const int*)  d_in[0];
    const float* hidden = (const float*)d_in[1];
    const float* emb    = (const float*)d_in[2];
    const float* Wr     = (const float*)d_in[3];
    const float* Ur     = (const float*)d_in[4];
    const float* br     = (const float*)d_in[5];
    const float* Wz     = (const float*)d_in[6];
    const float* Uz     = (const float*)d_in[7];
    const float* bz     = (const float*)d_in[8];
    const float* Wh     = (const float*)d_in[9];
    const float* Uh     = (const float*)d_in[10];
    const float* bh     = (const float*)d_in[11];
    const float* Wout   = (const float*)d_in[12];
    const float* bout   = (const float*)d_in[13];

    constexpr size_t HH = (size_t)H_ * H_;
    constexpr size_t BH = (size_t)B_ * H_;
    constexpr size_t WB = (size_t)L_ * HH * 2;      // bytes per bf16 weight tensor

    auto alignup = [](size_t x) { return (x + 255) & ~(size_t)255; };
    char*  w = (char*)d_ws;
    size_t o = 0;
    __bf16* bWr   = (__bf16*)(w + o); o = alignup(o + WB);
    __bf16* bUr   = (__bf16*)(w + o); o = alignup(o + WB);
    __bf16* bWz   = (__bf16*)(w + o); o = alignup(o + WB);
    __bf16* bUz   = (__bf16*)(w + o); o = alignup(o + WB);
    __bf16* bWh   = (__bf16*)(w + o); o = alignup(o + WB);
    __bf16* bUh   = (__bf16*)(w + o); o = alignup(o + WB);
    __bf16* bWout = (__bf16*)(w + o); o = alignup(o + (size_t)V_ * H_ * 2);
    __bf16* xe    = (__bf16*)(w + o); o = alignup(o + (size_t)T_ * B_ * E_ * 2);
    __bf16* tops  = (__bf16*)(w + o); o = alignup(o + (size_t)T_ * B_ * H_ * 2);
    float*  hfp   = (float*) (w + o); o = alignup(o + (size_t)L_ * BH * 4);
    __bf16* hbf   = (__bf16*)(w + o); o = alignup(o + (size_t)L_ * BH * 2);
    float*  zbuf  = (float*) (w + o); o = alignup(o + BH * 4);
    float*  cpart = (float*) (w + o); o = alignup(o + BH * 4);
    __bf16* rhbf  = (__bf16*)(w + o); o = alignup(o + BH * 2);

    // --- one-time conversions ---
    {
        size_t n = (size_t)L_ * HH;
        int g = (int)((n + 255) / 256);
        f2bf_kernel<<<g, 256, 0, stream>>>(Wr, bWr, n);
        f2bf_kernel<<<g, 256, 0, stream>>>(Ur, bUr, n);
        f2bf_kernel<<<g, 256, 0, stream>>>(Wz, bWz, n);
        f2bf_kernel<<<g, 256, 0, stream>>>(Uz, bUz, n);
        f2bf_kernel<<<g, 256, 0, stream>>>(Wh, bWh, n);
        f2bf_kernel<<<g, 256, 0, stream>>>(Uh, bUh, n);
        size_t nv = (size_t)V_ * H_;
        f2bf_kernel<<<(int)((nv + 255) / 256), 256, 0, stream>>>(Wout, bWout, nv);
    }
    {
        size_t n = (size_t)T_ * B_ * E_;
        embed_kernel<<<(int)((n + 255) / 256), 256, 0, stream>>>(inputs, emb, xe);
    }
    {
        int n = (int)((size_t)L_ * BH);
        inith_kernel<<<(n + 255) / 256, 256, 0, stream>>>(hidden, hfp, hbf, n);
    }

    // --- recurrence ---
    const int NBLK = H_ / 32;   // 32 blocks
    for (int t = 0; t < T_; ++t) {
        const __bf16* x0 = xe + (size_t)t * B_ * E_;
        // layer 0
        gru_cell_a<<<NBLK, 256, 0, stream>>>(
            x0, hbf, hfp,
            bWr, bWz, bWh, bUr, bUz,
            br, bz, zbuf, cpart, rhbf);
        gru_cell_b<<<NBLK, 256, 0, stream>>>(
            rhbf, bUh, bh, cpart, zbuf, hfp, hbf, (__bf16*)nullptr);
        // layer 1 (input = fresh h0 in bf16)
        gru_cell_a<<<NBLK, 256, 0, stream>>>(
            hbf, hbf + BH, hfp + BH,
            bWr + HH, bWz + HH, bWh + HH, bUr + HH, bUz + HH,
            br + H_, bz + H_, zbuf, cpart, rhbf);
        gru_cell_b<<<NBLK, 256, 0, stream>>>(
            rhbf, bUh + HH, bh + H_, cpart, zbuf, hfp + BH, hbf + BH,
            tops + (size_t)t * B_ * H_);
    }

    // --- output projection + final hidden ---
    dim3 lgrid((V_ + 63) / 64, (T_ * B_) / 128);
    logits_gemm<<<lgrid, 256, 0, stream>>>(tops, bWout, bout, (float*)d_out);

    {
        int n = (int)((size_t)L_ * BH);
        copyh_kernel<<<(n + 255) / 256, 256, 0, stream>>>(
            hfp, (float*)d_out + (size_t)T_ * B_ * V_, n);
    }
}
